// EPROP3_LSTM_53360673685594
// MI455X (gfx1250) — matrix operand order repaired and checked by reading, not compiled, not run on hardware
//
#include <hip/hip_runtime.h>

// ---------------------------------------------------------------------------
// MI455X (gfx1250) e-prop LSTM.
//
// Reformulation: final eligibility traces are linear in the per-step rank-1
// updates, so  ev_x[b] = beta[b](3H x T) @ X[b](T x I),
//              ev_h[b] = beta[b](3H x T) @ Hprev[b](T x H),
// where beta[b,j,t] = alpha[b,j,t] * prod_{t'>t} f[b, j%H, t'].
// All heavy math becomes bf16 WMMA GEMMs with f32 accumulation
// (v_wmma_f32_16x16x32_bf16); only the 256-step h-recurrence is sequential.
//
// GEMM kernel: 32x32 macro-tile per wave (2 A-frags x 2 B-frags -> 4 WMMAs
// per K-step of 32) for 2x the flops/byte of a single-tile wave; pointers
// strength-reduced across the K loop; unconditional speculative prefetch.
// ---------------------------------------------------------------------------

typedef __attribute__((ext_vector_type(16))) __bf16 v16bf;
typedef __attribute__((ext_vector_type(8)))  __bf16 v8bf;
typedef __attribute__((ext_vector_type(8)))  float  v8f;

#define B_  64
#define T_  256
#define I_  256
#define H_  512
#define O_  128
#define FH_ 2048   // 4H
#define TH_ 1536   // 3H

static __device__ __forceinline__ unsigned short f2bf(float f) {
  unsigned int u = __float_as_uint(f);
  unsigned int r = u + 0x7FFFu + ((u >> 16) & 1u);   // round-to-nearest-even
  return (unsigned short)(r >> 16);
}
static __device__ __forceinline__ float bf2f(unsigned short b) {
  return __uint_as_float(((unsigned int)b) << 16);
}

static __device__ __forceinline__ v16bf make_a(const __bf16* p) {
  v8bf lo = *(const v8bf*)(p);
  v8bf hi = *(const v8bf*)(p + 16);
  return __builtin_shufflevector(lo, hi, 0,1,2,3,4,5,6,7,8,9,10,11,12,13,14,15);
}

// ---------------------------------------------------------------------------
// Generic (optionally batched) bf16 WMMA GEMM:
//   C = A(bf16) x B(bf16) [+ Cadd] [+ bias]
// A: M x K row-major (lda), B: K x N row-major (ldb), C: M x N f32 (ldc).
// One 32x32 macro-tile per wave; 8 waves / 256-thread block.
// Grid must exactly cover tiles (EXEC all-ones for WMMA).
// ---------------------------------------------------------------------------
__global__ void __launch_bounds__(256) wmma_gemm_bf16(
    const unsigned short* __restrict__ A, long long lda, long long strideA,
    const unsigned short* __restrict__ Bm, long long ldb, long long strideB,
    float* __restrict__ C, long long ldc, long long strideC,
    const float* __restrict__ Cadd, long long ldcadd,
    const float* __restrict__ bias,
    int Mtiles, int Ntiles, int K)            // tiles are 32x32
{
  const int lane = threadIdx.x & 31;
  const int wave = threadIdx.x >> 5;
  const long long tile = (long long)blockIdx.x * 8 + wave;
  const int tilesPB = Mtiles * Ntiles;
  const int batch = (int)(tile / tilesPB);
  const int rem   = (int)(tile % tilesPB);
  const int tm = rem / Ntiles;
  const int tn = rem % Ntiles;

  const __bf16* Ab = (const __bf16*)A + (long long)batch * strideA + (long long)(tm * 32) * lda;
  const __bf16* Bb = (const __bf16*)Bm + (long long)batch * strideB + (long long)(tn * 32);

  // A-frag addressing per ISA 16-bit 16x32 layout: row m = lane&15,
  // k-chunks {koff..koff+7} and {koff+16..koff+23}, koff = 8*(lane>>4).
  const int m    = lane & 15;
  const int koff = (lane >> 4) * 8;
  const __bf16* pa0 = Ab + (long long)m * lda + koff;          // rows tm*32 + m
  const __bf16* pa1 = pa0 + 16 * lda;                          // rows tm*32 + 16 + m
  // B-frag: column n = lane&15; k-chunks {koff..koff+7} and {koff+16..koff+23}.
  const __bf16* pb  = Bb + (long long)koff * ldb + (lane & 15);
  const long long ldbStep = 32 * ldb;

  v8f acc00 = {}, acc01 = {}, acc10 = {}, acc11 = {};
  for (int k0 = 0; k0 < K; k0 += 32) {
    v16bf a0 = make_a(pa0);
    v16bf a1 = make_a(pa1);
    v16bf b0, b1;
#pragma unroll
    for (int s = 0; s < 16; ++s) {
      const long long kk = (long long)((s & 7) + (s >> 3) * 16) * ldb;
      b0[s] = pb[kk];
      b1[s] = pb[kk + 16];
    }

    // speculative streaming prefetch of next K block (global_prefetch_b8)
    __builtin_prefetch(pa0 + 32, 0, 0);
    __builtin_prefetch(pa1 + 32, 0, 0);
    __builtin_prefetch(pb + ldbStep, 0, 0);

    acc00 = __builtin_amdgcn_wmma_f32_16x16x32_bf16(false, a0, false, b0, (short)0, acc00, false, false);
    acc01 = __builtin_amdgcn_wmma_f32_16x16x32_bf16(false, a0, false, b1, (short)0, acc01, false, false);
    acc10 = __builtin_amdgcn_wmma_f32_16x16x32_bf16(false, a1, false, b0, (short)0, acc10, false, false);
    acc11 = __builtin_amdgcn_wmma_f32_16x16x32_bf16(false, a1, false, b1, (short)0, acc11, false, false);

    pa0 += 32;
    pa1 += 32;
    pb  += ldbStep;
  }

  // C/D layout: VGPR r -> M = r + 8*(lane>>4), N = lane&15.
  float* Cb = C + (long long)batch * strideC;
  const int n  = lane & 15;
  const int mh = (lane >> 4) * 8;
  const long long row0 = (long long)tm * 32;
  const long long col0 = (long long)tn * 32 + n;
  const float bias0 = bias ? bias[col0]      : 0.0f;
  const float bias1 = bias ? bias[col0 + 16] : 0.0f;

  v8f accs[4] = {acc00, acc01, acc10, acc11};
#pragma unroll
  for (int q = 0; q < 4; ++q) {
    const int dm = (q >> 1) * 16;
    const long long col = col0 + (q & 1) * 16;
    const float bv = (q & 1) ? bias1 : bias0;
#pragma unroll
    for (int r = 0; r < 8; ++r) {
      long long row = row0 + dm + mh + r;
      float v = accs[q][r] + bv;
      if (Cadd) v += Cadd[row * ldcadd + col];
      Cb[row * ldc + col] = v;
    }
  }
}

// ---------------------------------------------------------------------------
// Per-step gate nonlinearity + state update + alpha (written in beta layout).
// 32768 threads: tid -> (b, hh).
// ---------------------------------------------------------------------------
__global__ void __launch_bounds__(256) lstm_gates(
    const float* __restrict__ z, float* __restrict__ c,
    unsigned short* __restrict__ h_bf,           // (B, T+1, H) bf16, row t+1 written
    float* __restrict__ f_all,                   // (B, T, H) f32
    unsigned short* __restrict__ beta_bf,        // (B, 3H, T) bf16 (alpha for now)
    int t)
{
  const int tid = blockIdx.x * blockDim.x + threadIdx.x;
  const int b  = tid >> 9;
  const int hh = tid & 511;
  const float* zr = z + (long long)b * FH_;
  const float zi = zr[hh];
  const float zf = zr[H_ + hh];
  const float zg = zr[2 * H_ + hh];
  const float zo = zr[3 * H_ + hh];
  const float ig = 1.0f / (1.0f + __expf(-zi));
  const float fg = 1.0f / (1.0f + __expf(-zf));
  const float gg = tanhf(zg);
  const float og = 1.0f / (1.0f + __expf(-zo));
  const float cold = c[tid];
  const float cnew = fg * cold + ig * gg;
  const float hnew = og * tanhf(cnew);
  c[tid] = cnew;
  h_bf[((long long)b * (T_ + 1) + (t + 1)) * H_ + hh] = f2bf(hnew);
  f_all[((long long)b * T_ + t) * H_ + hh] = fg;
  const float d_i = gg * ig * (1.0f - ig);
  const float d_f = cold * fg * (1.0f - fg);
  const float d_g = ig * (1.0f - gg * gg);
  unsigned short* bb = beta_bf + (long long)b * TH_ * T_ + t;
  bb[(long long)hh * T_]            = f2bf(d_i);
  bb[(long long)(H_ + hh) * T_]     = f2bf(d_f);
  bb[(long long)(2 * H_ + hh) * T_] = f2bf(d_g);
}

// ---------------------------------------------------------------------------
// Suffix products of f: scale alpha -> beta in place, reduce ev_b.
// beta[b,j,t] *= prod_{t'>t} f[b, j%H, t'] ;  ev_b[b,j] = sum_t beta.
// ---------------------------------------------------------------------------
__global__ void __launch_bounds__(256) suffix_scale(
    const float* __restrict__ f_all, unsigned short* __restrict__ beta_bf,
    float* __restrict__ evb_out)
{
  const int tid = blockIdx.x * blockDim.x + threadIdx.x;
  const int b = tid >> 9;
  const int hh = tid & 511;
  unsigned short* b0 = beta_bf + (long long)b * TH_ * T_ + (long long)hh * T_;
  unsigned short* b1 = b0 + (long long)H_ * T_;
  unsigned short* b2 = b0 + (long long)(2 * H_) * T_;
  const float* fr = f_all + (long long)b * T_ * H_ + hh;
  float P = 1.0f, s0 = 0.0f, s1 = 0.0f, s2 = 0.0f;
  for (int t = T_ - 1; t >= 0; --t) {
    float a0 = bf2f(b0[t]) * P;
    float a1 = bf2f(b1[t]) * P;
    float a2 = bf2f(b2[t]) * P;
    b0[t] = f2bf(a0); b1[t] = f2bf(a1); b2[t] = f2bf(a2);
    s0 += a0; s1 += a1; s2 += a2;
    P *= fr[(long long)t * H_];
  }
  evb_out[(long long)b * TH_ + hh]            = s0;
  evb_out[(long long)b * TH_ + H_ + hh]       = s1;
  evb_out[(long long)b * TH_ + 2 * H_ + hh]   = s2;
}

// Small f32 dense layer for the 64-row MLP head (negligible FLOPs).
__global__ void __launch_bounds__(256) dense_layer(
    const float* __restrict__ in, long long in_stride,
    const float* __restrict__ W, const float* __restrict__ bias,
    float* __restrict__ out, long long out_stride, int K, int N, int relu)
{
  const int tid = blockIdx.x * blockDim.x + threadIdx.x;
  const int b = tid / N;
  const int n = tid % N;
  const float* ir = in + (long long)b * in_stride;
  const float* wr = W + (long long)n * K;
  float s = bias[n];
  for (int k = 0; k < K; ++k) s = fmaf(ir[k], wr[k], s);
  if (relu) s = fmaxf(s, 0.0f);
  out[(long long)b * out_stride + n] = s;
}

__global__ void __launch_bounds__(256) init_state(
    float* __restrict__ cbuf, unsigned short* __restrict__ h_bf,
    float* __restrict__ old_c)
{
  const int tid = blockIdx.x * blockDim.x + threadIdx.x;  // 32768
  const int b = tid >> 9;
  const int hh = tid & 511;
  cbuf[tid] = 0.0f;
  old_c[tid] = 0.0f;
  h_bf[(long long)b * (T_ + 1) * H_ + hh] = 0;  // h_{-1} = 0
}

__global__ void __launch_bounds__(256) conv_bf16(
    const float* __restrict__ in, unsigned short* __restrict__ out, long long n)
{
  const long long tid = (long long)blockIdx.x * blockDim.x + threadIdx.x;
  if (tid < n) out[tid] = f2bf(in[tid]);
}

// out[(c, r)] = bf16(in[r, c]) : (R x C) -> (C x R)
__global__ void __launch_bounds__(256) transpose_bf16(
    const float* __restrict__ in, unsigned short* __restrict__ out, int R, int C)
{
  const int tid = blockIdx.x * blockDim.x + threadIdx.x;
  if (tid < R * C) {
    const int r = tid / C, c = tid % C;
    out[(long long)c * R + r] = f2bf(in[tid]);
  }
}

extern "C" void kernel_launch(void* const* d_in, const int* in_sizes, int n_in,
                              void* d_out, int out_size, void* d_ws, size_t ws_size,
                              hipStream_t stream) {
  (void)in_sizes; (void)n_in; (void)out_size; (void)ws_size;
  const float* x    = (const float*)d_in[0];
  const float* W_ih = (const float*)d_in[1];
  const float* W_hh = (const float*)d_in[2];
  const float* b_l  = (const float*)d_in[3];
  const float* W_d  = (const float*)d_in[4];
  const float* b_d  = (const float*)d_in[5];
  const float* W1   = (const float*)d_in[6];
  const float* b1   = (const float*)d_in[7];
  const float* W2   = (const float*)d_in[8];
  const float* b2   = (const float*)d_in[9];
  const float* W3   = (const float*)d_in[10];
  const float* b3   = (const float*)d_in[11];
  const float* W4   = (const float*)d_in[12];
  const float* b4   = (const float*)d_in[13];

  float* out      = (float*)d_out;
  float* out_lstm = out;                               // (B,T,O)
  float* out_sg   = out + 2097152;                     // (B,H)
  float* out_oc   = out + 2129920;                     // (B,H) zeros
  float* out_evx  = out + 2162688;                     // (B,3H,I)
  float* out_evh  = out + 27328512;                    // (B,3H,H)
  float* out_evb  = out + 77660160;                    // (B,3H,1)

  char* ws = (char*)d_ws;
  unsigned short* x_bf    = (unsigned short*)(ws + 0);              //  8.0 MB
  unsigned short* WihT    = (unsigned short*)(ws + 8388608);        //  1.0 MB (I x 4H)
  unsigned short* WhhT    = (unsigned short*)(ws + 9437184);        //  2.0 MB (H x 4H)
  unsigned short* WdT     = (unsigned short*)(ws + 11534336);       //  128 KB (H x O)
  unsigned short* h_bf    = (unsigned short*)(ws + 11665408);       // 16.8 MB (B, T+1, H)
  float*          cbuf    = (float*)(ws + 28508160);                //  128 KB
  float*          zbuf    = (float*)(ws + 28639232);                //  512 KB (B, 4H)
  float*          s1      = (float*)(ws + 29163520);
  float*          s2      = (float*)(ws + 29294592);
  float*          s3      = (float*)(ws + 29360128);
  float*          f_all   = (float*)(ws + 29392896);                // 33.6 MB (B,T,H)
  unsigned short* beta_bf = (unsigned short*)(ws + 62947328);       // 50.3 MB (B,3H,T)
  float*          Xp      = (float*)(ws + 113278976);               // 134 MB (B*T, 4H)

  // ---- init + precision conversion / weight transposes -------------------
  init_state<<<128, 256, 0, stream>>>(cbuf, h_bf, out_oc);
  conv_bf16<<<16384, 256, 0, stream>>>(x, x_bf, (long long)B_ * T_ * I_);
  transpose_bf16<<<2048, 256, 0, stream>>>(W_ih, WihT, FH_, I_);
  transpose_bf16<<<4096, 256, 0, stream>>>(W_hh, WhhT, FH_, H_);
  transpose_bf16<<<256,  256, 0, stream>>>(W_d,  WdT,  O_,  H_);

  // ---- Xp = x @ W_ih^T + b_lstm : (16384 x 2048 x 256), fully parallel ---
  wmma_gemm_bf16<<<4096, 256, 0, stream>>>(
      x_bf, I_, 0, WihT, FH_, 0, Xp, FH_, 0,
      nullptr, 0, b_l, /*Mtiles=*/512, /*Ntiles=*/64, /*K=*/I_);

  // ---- sequential scan: z = Xp_t + h_{t-1} @ W_hh^T, then gates ----------
  for (int t = 0; t < T_; ++t) {
    wmma_gemm_bf16<<<16, 256, 0, stream>>>(
        h_bf + (long long)t * H_, (long long)(T_ + 1) * H_, 0,
        WhhT, FH_, 0,
        zbuf, FH_, 0,
        Xp + (long long)t * FH_, (long long)T_ * FH_,
        nullptr, /*Mtiles=*/2, /*Ntiles=*/64, /*K=*/H_);
    lstm_gates<<<128, 256, 0, stream>>>(zbuf, cbuf, h_bf, f_all, beta_bf, t);
  }

  // ---- beta = alpha * suffix-prod(f); ev_b reduction ---------------------
  suffix_scale<<<128, 256, 0, stream>>>(f_all, beta_bf, out_evb);

  // ---- ev_x[b] = beta[b] @ X[b] : 64 x (1536 x 256 x 256) ----------------
  wmma_gemm_bf16<<<3072, 256, 0, stream>>>(
      beta_bf, T_, (long long)TH_ * T_,
      x_bf, I_, (long long)T_ * I_,
      out_evx, I_, (long long)TH_ * I_,
      nullptr, 0, nullptr, /*Mtiles=*/48, /*Ntiles=*/8, /*K=*/T_);

  // ---- ev_h[b] = beta[b] @ Hprev[b] : 64 x (1536 x 512 x 256) ------------
  wmma_gemm_bf16<<<6144, 256, 0, stream>>>(
      beta_bf, T_, (long long)TH_ * T_,
      h_bf, H_, (long long)(T_ + 1) * H_,          // rows t hold h_{t-1}
      out_evh, H_, (long long)TH_ * H_,
      nullptr, 0, nullptr, /*Mtiles=*/48, /*Ntiles=*/16, /*K=*/T_);

  // ---- lstm_out = hs @ W_d^T + b_d : batched (256 x 128 x 512) x 64 ------
  wmma_gemm_bf16<<<256, 256, 0, stream>>>(
      h_bf + H_, H_, (long long)(T_ + 1) * H_,     // rows = h_t
      WdT, O_, 0,
      out_lstm, O_, (long long)T_ * O_,
      nullptr, 0, b_d, /*Mtiles=*/8, /*Ntiles=*/4, /*K=*/H_);

  // ---- MLP head on last timestep (tiny) ----------------------------------
  dense_layer<<<128, 256, 0, stream>>>(out_lstm + (long long)(T_ - 1) * O_,
                                       (long long)T_ * O_, W1, b1, s1, 512, O_, 512, 1);
  dense_layer<<<64, 256, 0, stream>>>(s1, 512, W2, b2, s2, 256, 512, 256, 1);
  dense_layer<<<32, 256, 0, stream>>>(s2, 256, W3, b3, s3, 128, 256, 128, 1);
  dense_layer<<<128, 256, 0, stream>>>(s3, 128, W4, b4, out_sg, 512, 128, 512, 0);
}
